// EquiformerV2Block_36807869727018
// MI455X (gfx1250) — matrix operand order repaired and checked by reading, not compiled, not run on hardware
//
#include <hip/hip_runtime.h>
#include <cstdint>
#include <cstddef>

typedef __attribute__((ext_vector_type(16))) _Float16 v16h;
typedef __attribute__((ext_vector_type(8)))  _Float16 v8h;
typedef __attribute__((ext_vector_type(8)))  float    v8f;

#define DEV __device__ __forceinline__

static constexpr int SK   = 25;   // (LMAX+1)^2
static constexpr int SKR  = 19;
static constexpr int SC   = 64;

// f16 weight buffer offsets (in halves). All matmul weights stored TRANSPOSED
// (column-major: [n][k], ld = K) so B fragments are contiguous per lane.
static constexpr int OFF_WE1T  = 0;        // [128 n][128 k]
static constexpr int OFF_WE2T  = 16384;    // [64 n][128 k]
static constexpr int OFF_WSO2T = 24576;    // [64 n][128 k] (k 0-63 top, 64-127 bottom)
static constexpr int OFF_WVT   = 32768;    // [64 n][64 k]
static constexpr int OFF_WOT   = 36864;    // [64 n][64 k]
static constexpr int OFF_WF1T  = 40960;    // [128 n][64 k]
static constexpr int OFF_WF2T  = 49152;    // [64 n][128 k]
static constexpr int OFF_TG    = 57344;    // [128][32] row-major zero-padded to_grid (100x25)
static constexpr int OFF_FG    = 61440;    // [32][128] row-major zero-padded from_grid (25x100)
static constexpr int WB_TOTAL  = 65536;

DEV v8f wmma(v16h a, v16h b, v8f c) {
  return __builtin_amdgcn_wmma_f32_16x16x32_f16(false, a, false, b, (short)0, c, false, false);
}

// ---- fragment helpers (ISA 7.12.2 layouts, wave32) ----

// A fragment 16x32 from f16 row-major, leading dim ld (halves). 16B-aligned rows.
DEV v16h frag_a_h(const _Float16* base, int ld) {
  int lane = threadIdx.x & 31;
  int m = lane & 15;
  int klo = (lane < 16) ? 0 : 8;
  const _Float16* p = base + m * ld + klo;
  v8h lo = *(const v8h*)p;
  v8h hi = *(const v8h*)(p + 16);
  return __builtin_shufflevector(lo, hi, 0,1,2,3,4,5,6,7,8,9,10,11,12,13,14,15);
}

// A fragment with row bound (zero-fill)
DEV v16h frag_a_hb(const _Float16* base, int ld, int validRows) {
  int lane = threadIdx.x & 31;
  int m = lane & 15;
  int klo = (lane < 16) ? 0 : 8;
  if (m < validRows) {
    const _Float16* p = base + m * ld + klo;
    v8h lo = *(const v8h*)p;
    v8h hi = *(const v8h*)(p + 16);
    return __builtin_shufflevector(lo, hi, 0,1,2,3,4,5,6,7,8,9,10,11,12,13,14,15);
  }
  v16h r;
#pragma unroll
  for (int i = 0; i < 16; ++i) r[i] = (_Float16)0.0f;
  return r;
}

// A fragment from f32 source (LDS-staged Wigner block), row+col bounds, zero-pad
DEV v16h frag_a_f32b(const float* base, int mt, int rows, int cols, int ld) {
  int lane = threadIdx.x & 31;
  int m = mt * 16 + (lane & 15);
  int klo = (lane < 16) ? 0 : 8;
  v16h r;
#pragma unroll
  for (int i = 0; i < 8; ++i) {
    int k0 = klo + i, k1 = klo + 16 + i;
    r[i]     = (m < rows && k0 < cols) ? (_Float16)base[m * ld + k0] : (_Float16)0.0f;
    r[8 + i] = (m < rows && k1 < cols) ? (_Float16)base[m * ld + k1] : (_Float16)0.0f;
  }
  return r;
}

// B fragment 32x16 from COLUMN-MAJOR f16 [n][k], leading dim ldk (halves).
// Per lane: k = koff..koff+7 and koff+16..koff+23 -> two 16B loads.
DEV v16h frag_b_cm(const _Float16* base, int ldk) {
  int lane = threadIdx.x & 31;
  int n = lane & 15;
  int koff = (lane < 16) ? 0 : 8;
  const _Float16* p = base + n * ldk + koff;
  v8h lo = *(const v8h*)p;
  v8h hi = *(const v8h*)(p + 16);
  return __builtin_shufflevector(lo, hi, 0,1,2,3,4,5,6,7,8,9,10,11,12,13,14,15);
}

// D store (f32 acc -> f16 buffer), row-major, zero-fill rows >= validRows
DEV void store_d_fill(v8f d, _Float16* base, int ld, int validRows) {
  int lane = threadIdx.x & 31;
  int n = lane & 15;
  int m0 = (lane < 16) ? 0 : 8;
#pragma unroll
  for (int i = 0; i < 8; ++i) {
    int m = m0 + i;
    base[m * ld + n] = (m < validRows) ? (_Float16)d[i] : (_Float16)0.0f;
  }
}

// D store into COLUMN-MAJOR f16 buffer (ldk = rows). One 16B store per lane.
DEV void store_d_cm(v8f d, _Float16* base, int ldk, int validRows) {
  int lane = threadIdx.x & 31;
  int n = lane & 15;
  int m0 = (lane < 16) ? 0 : 8;
  v8h o;
#pragma unroll
  for (int i = 0; i < 8; ++i) {
    int m = m0 + i;
    o[i] = (m < validRows) ? (_Float16)d[i] : (_Float16)0.0f;
  }
  *(v8h*)(base + n * ldk + m0) = o;
}

// D store to f16 global, row-major; fast path when all rows valid
DEV void store_d_skip_h(v8f d, _Float16* base, int ld, int validRows) {
  int lane = threadIdx.x & 31;
  int n = lane & 15;
  int m0 = (lane < 16) ? 0 : 8;
  if (validRows >= 16) {
#pragma unroll
    for (int i = 0; i < 8; ++i) base[(m0 + i) * ld + n] = (_Float16)d[i];
  } else {
#pragma unroll
    for (int i = 0; i < 8; ++i) { int m = m0 + i; if (m < validRows) base[m * ld + n] = (_Float16)d[i]; }
  }
}

// D store to f32 global, row-major; fast path when all rows valid
DEV void store_d_skip_f(v8f d, float* base, int ld, int validRows) {
  int lane = threadIdx.x & 31;
  int n = lane & 15;
  int m0 = (lane < 16) ? 0 : 8;
  if (validRows >= 16) {
#pragma unroll
    for (int i = 0; i < 8; ++i) base[(m0 + i) * ld + n] = d[i];
  } else {
#pragma unroll
    for (int i = 0; i < 8; ++i) { int m = m0 + i; if (m < validRows) base[m * ld + n] = d[i]; }
  }
}

DEV float sigm(float x) { return 1.0f / (1.0f + __expf(-x)); }

DEV unsigned f2ord(float f) { unsigned u = __float_as_uint(f); return (u & 0x80000000u) ? (~u) : (u | 0x80000000u); }
DEV float    ord2f(unsigned u) { unsigned b = (u & 0x80000000u) ? (u ^ 0x80000000u) : (~u); return __uint_as_float(b); }

// async copy of one b128 per lane: global -> LDS (gfx1250 async path, ASYNCcnt)
DEV void async_g2l_b128(unsigned lds_off, const void* gaddr) {
  asm volatile("global_load_async_to_lds_b128 %0, %1, off"
               :: "v"(lds_off), "v"((unsigned long long)(uintptr_t)gaddr)
               : "memory");
}
DEV void wait_asynccnt0() {
  asm volatile("s_wait_asynccnt 0x0" ::: "memory");
}

// ================== kernels ==================

// residual init + softmax accumulator init
__global__ void k_init(const float* __restrict__ nf, float* __restrict__ x1,
                       float* __restrict__ den, unsigned* __restrict__ mxo,
                       long long nkc, long long n4) {
  long long i = (long long)blockIdx.x * blockDim.x + threadIdx.x;
  long long st = (long long)gridDim.x * blockDim.x;
  for (long long t = i; t < nkc; t += st) x1[t] = nf[t];
  for (long long t = i; t < n4; t += st) { den[t] = 0.0f; mxo[t] = 0u; }
}

// weight conversion to f16; matmul weights transposed to column-major
__global__ void k_cvt(const float* We1, const float* We2, const float* Wso2,
                      const float* Wv, const float* Wo, const float* Wf1, const float* Wf2,
                      const float* tg, const float* fg, _Float16* wb) {
  int i = blockIdx.x * blockDim.x + threadIdx.x;
  int st = gridDim.x * blockDim.x;
  for (int t = i; t < WB_TOTAL; t += st) {
    float v;
    if (t < OFF_WE2T)        { int j = t;             int n = j >> 7, k = j & 127; v = We1[k * 128 + n]; }
    else if (t < OFF_WSO2T)  { int j = t - OFF_WE2T;  int n = j >> 7, k = j & 127; v = We2[k * 64 + n]; }
    else if (t < OFF_WVT)    { int j = t - OFF_WSO2T; int n = j >> 7, k = j & 127; v = Wso2[k * 64 + n]; }
    else if (t < OFF_WOT)    { int j = t - OFF_WVT;   int n = j >> 6, k = j & 63;  v = Wv[k * 64 + n]; }
    else if (t < OFF_WF1T)   { int j = t - OFF_WOT;   int n = j >> 6, k = j & 63;  v = Wo[k * 64 + n]; }
    else if (t < OFF_WF2T)   { int j = t - OFF_WF1T;  int n = j >> 6, k = j & 63;  v = Wf1[k * 128 + n]; }
    else if (t < OFF_TG)     { int j = t - OFF_WF2T;  int n = j >> 7, k = j & 127; v = Wf2[k * 64 + n]; }
    else if (t < OFF_FG)     { int j = t - OFF_TG;    int rr = j >> 5, cc = j & 31;
                               v = (rr < 100 && cc < SK) ? tg[rr * SK + cc] : 0.0f; }
    else                     { int j = t - OFF_FG;    int rr = j >> 7, cc = j & 127;
                               v = (rr < SK && cc < 100) ? fg[rr * 100 + cc] : 0.0f; }
    wb[t] = (_Float16)v;
  }
}

// per-node RMS norm over K*C, per-degree gamma, write f16
__global__ void k_norm(const float* __restrict__ xin, const float* __restrict__ g,
                       _Float16* __restrict__ xout) {
  int n = blockIdx.x;
  const float* x = xin + (size_t)n * SK * SC;
  __shared__ float red[256];
  float s = 0.0f;
  for (int i = threadIdx.x; i < SK * SC; i += blockDim.x) { float v = x[i]; s += v * v; }
  red[threadIdx.x] = s;
  __syncthreads();
  for (int o = 128; o > 0; o >>= 1) {
    if (threadIdx.x < o) red[threadIdx.x] += red[threadIdx.x + o];
    __syncthreads();
  }
  float inv = 1.0f / sqrtf(red[0] / (float)(SK * SC) + 1e-6f);
  for (int i = threadIdx.x; i < SK * SC; i += blockDim.x) {
    int k = i >> 6, c = i & 63;
    int l = (k < 1) ? 0 : (k < 4) ? 1 : (k < 9) ? 2 : (k < 16) ? 3 : 4;
    xout[(size_t)n * SK * SC + i] = (_Float16)(x[i] * inv * g[l * 64 + c]);
  }
}

// y = xh @ Wso2_top, z = xh @ Wso2_bot  (M = N*K rows, K=64, N=64)  [WMMA]
__global__ void k_yz(const _Float16* __restrict__ xh, const _Float16* __restrict__ wb,
                     _Float16* __restrict__ yh, _Float16* __restrict__ zh, int M) {
  int wv = (int)((blockIdx.x * blockDim.x + threadIdx.x) >> 5);
  int row0 = wv * 16;
  if (row0 >= M) return;
  int vr = M - row0; if (vr > 16) vr = 16;
  v16h a0 = frag_a_hb(xh + (size_t)row0 * 64, 64, vr);
  v16h a1 = frag_a_hb(xh + (size_t)row0 * 64 + 32, 64, vr);
  const _Float16* Wt = wb + OFF_WSO2T;  // [64 n][128 k]
#pragma unroll
  for (int nt = 0; nt < 4; ++nt) {
    v8f accy = {}; v8f accz = {};
    accy = wmma(a0, frag_b_cm(Wt + (nt * 16) * 128 + 0, 128), accy);
    accy = wmma(a1, frag_b_cm(Wt + (nt * 16) * 128 + 32, 128), accy);
    accz = wmma(a0, frag_b_cm(Wt + (nt * 16) * 128 + 64, 128), accz);
    accz = wmma(a1, frag_b_cm(Wt + (nt * 16) * 128 + 96, 128), accz);
    store_d_skip_h(accy, yh + (size_t)row0 * 64 + nt * 16, 64, vr);
    store_d_skip_h(accz, zh + (size_t)row0 * 64 + nt * 16, 64, vr);
  }
}

// edge embedding MLP: [E,128] -> silu -> @We2 -> silu -> ef[E,64]  [WMMA]
__global__ __launch_bounds__(128) void k_edgemlp(
    const float* __restrict__ edist, const int* __restrict__ spec,
    const int* __restrict__ snd, const int* __restrict__ rcv,
    const float* __restrict__ embs, const float* __restrict__ embr,
    const _Float16* __restrict__ wb, float* __restrict__ ef, int E) {
  __shared__ __align__(16) _Float16 s_in[4][2048];
  __shared__ __align__(16) _Float16 s_h1[4][2048];
  int w = threadIdx.x >> 5, lane = threadIdx.x & 31;
  int e0 = (blockIdx.x * 4 + w) * 16;
  if (e0 >= E) return;
  for (int idx = lane; idx < 2048; idx += 32) {
    int r = idx >> 7, c = idx & 127;
    int e = e0 + r;
    float v = 0.0f;
    if (e < E) {
      if (c < 64)      v = edist[(size_t)e * 64 + c];
      else if (c < 96) { int sp = spec[snd[e]]; v = embs[sp * 32 + (c - 64)]; }
      else             { int sp = spec[rcv[e]]; v = embr[sp * 32 + (c - 96)]; }
    }
    s_in[w][idx] = (_Float16)v;
  }
  const _Float16* W1 = wb + OFF_WE1T;  // [128 n][128 k]
#pragma unroll
  for (int nt = 0; nt < 8; ++nt) {
    v8f acc = {};
#pragma unroll
    for (int ks = 0; ks < 4; ++ks)
      acc = wmma(frag_a_h(&s_in[w][ks * 32], 128),
                 frag_b_cm(W1 + (nt * 16) * 128 + ks * 32, 128), acc);
#pragma unroll
    for (int i = 0; i < 8; ++i) { float x = acc[i]; acc[i] = x * sigm(x); }
    store_d_fill(acc, &s_h1[w][nt * 16], 128, 16);
  }
  const _Float16* W2 = wb + OFF_WE2T;  // [64 n][128 k]
  int vr = E - e0; if (vr > 16) vr = 16;
#pragma unroll
  for (int nt = 0; nt < 4; ++nt) {
    v8f acc = {};
#pragma unroll
    for (int ks = 0; ks < 4; ++ks)
      acc = wmma(frag_a_h(&s_h1[w][ks * 32], 128),
                 frag_b_cm(W2 + (nt * 16) * 128 + ks * 32, 128), acc);
#pragma unroll
    for (int i = 0; i < 8; ++i) { float x = acc[i]; acc[i] = x * sigm(x); }
    store_d_skip_f(acc, ef + (size_t)e0 * 64 + nt * 16, 64, vr);
  }
}

// attention logits from msg row 0 only; atomic segment max (ordered-uint)
__global__ __launch_bounds__(128) void k_logits(
    const float* __restrict__ wig, const _Float16* __restrict__ yh, const _Float16* __restrict__ zh,
    const float* __restrict__ ef, const float* __restrict__ Wa, const float* __restrict__ va,
    const int* __restrict__ snd, const int* __restrict__ rcv,
    float* __restrict__ logits, unsigned* __restrict__ mxo, int E) {
  __shared__ float g0s[4][64];
  int w = threadIdx.x >> 5, lane = threadIdx.x & 31;
  int e = blockIdx.x * 4 + w;
  if (e >= E) return;
  int s = snd[e], r = rcv[e];
  const float* wg = wig + (size_t)e * SKR * SK;  // row 0 of wigner
  int d0 = lane, d1 = lane + 32;
  float a0 = 0.0f, a1 = 0.0f;
  for (int k = 0; k < SK; ++k) {
    float wv = wg[k];
    size_t yb = ((size_t)s * SK + k) * 64, zb = ((size_t)r * SK + k) * 64;
    a0 += wv * ((float)yh[yb + d0] + (float)zh[zb + d0]);
    a1 += wv * ((float)yh[yb + d1] + (float)zh[zb + d1]);
  }
  float m0 = a0 * ef[(size_t)e * 64 + d0];
  float m1 = a1 * ef[(size_t)e * 64 + d1];
  g0s[w][d0] = m0 * sigm(m0);
  g0s[w][d1] = m1 * sigm(m1);
#pragma unroll
  for (int t = 0; t < 4; ++t) {
    int j = t * 32 + lane;
    float aj = 0.0f;
    for (int k = 0; k < 64; ++k) aj += g0s[w][k] * Wa[k * 128 + j];
    aj = (aj > 0.0f) ? aj : 0.2f * aj;
    float part = aj * va[t * 32 + lane];
    for (int off = 16; off > 0; off >>= 1) part += __shfl_xor(part, off, 32);
    if (lane == 0) {
      logits[(size_t)e * 4 + t] = part;
      atomicMax(&mxo[(size_t)r * 4 + t], f2ord(part));
    }
  }
}

// exp(logit - segmax), atomic segment sum
__global__ void k_expsum(const float* __restrict__ logits, const unsigned* __restrict__ mxo,
                         const int* __restrict__ rcv, float* __restrict__ exb,
                         float* __restrict__ den, int E) {
  int i = blockIdx.x * blockDim.x + threadIdx.x;
  if (i >= E * 4) return;
  int e = i >> 2, h = i & 3;
  int r = rcv[e];
  float m = ord2f(mxo[(size_t)r * 4 + h]);
  float v = __expf(logits[i] - m);
  exb[i] = v;
  atomicAdd(&den[(size_t)r * 4 + h], v);
}

// fused per-edge attention body. Wigner matrices prefetched via async global->LDS.
__global__ __launch_bounds__(128) void k_edge(
    const float* __restrict__ wig, const float* __restrict__ wiginv,
    const _Float16* __restrict__ yh, const _Float16* __restrict__ zh,
    const float* __restrict__ ef, const float* __restrict__ exb, const float* __restrict__ den,
    const int* __restrict__ snd, const int* __restrict__ rcv,
    const _Float16* __restrict__ wb, float* __restrict__ x1, int E) {
  __shared__ __align__(16) _Float16 b1[4][2048];  // xw (col-major), later u (col-major)
  __shared__ __align__(16) _Float16 b2[4][2048];  // msg (row-major)
  __shared__ __align__(16) _Float16 b3[4][2048];  // wigner staging (f32), later wv (row-major)
  __shared__ __align__(16) float    swg[4][512];  // wigner_inv staging (f32)
  __shared__ float gate[4][64];
  int w = threadIdx.x >> 5, lane = threadIdx.x & 31;
  int e = blockIdx.x * 4 + w;
  if (e >= E) return;
  int s = snd[e], r = rcv[e];
  _Float16* B1 = b1[w]; _Float16* B2 = b2[w]; _Float16* B3 = b3[w];
  float* sWig = (float*)B3;
  float* sWgi = swg[w];

  // 0) async prefetch wigner + wigner_inv (475 f32 = 1900B each) into LDS
  {
    unsigned lW = (unsigned)(uintptr_t)(void*)sWig;
    unsigned lI = (unsigned)(uintptr_t)(void*)sWgi;
    const char* gW = (const char*)(wig + (size_t)e * (SKR * SK));
    const char* gI = (const char*)(wiginv + (size_t)e * (SK * SKR));
#pragma unroll
    for (int t = 0; t < 4; ++t) {
      int off = (lane + t * 32) * 16;
      if (off > 1884) off = 1884;  // clamp: stay inside this edge's 1900B block
      async_g2l_b128(lW + off, gW + off);
      async_g2l_b128(lI + off, gI + off);
    }
  }

  // 1) xw = y[s] + z[r] -> B1 column-major [c][rr], rows >= 25 zero (overlaps async)
  for (int idx = lane; idx < 2048; idx += 32) {
    int c = idx >> 5, rr = idx & 31;
    float v = 0.0f;
    if (rr < SK) {
      v = (float)yh[((size_t)s * SK + rr) * 64 + c] + (float)zh[((size_t)r * SK + rr) * 64 + c];
    }
    B1[c * 32 + rr] = (_Float16)v;
  }

  wait_asynccnt0();  // wigner blocks now in LDS

  // 2) msg = (wigner @ xw) * ef -> B2 row-major (rows >= 19 zeroed)
#pragma unroll
  for (int mt = 0; mt < 2; ++mt) {
    v16h a = frag_a_f32b(sWig, mt, SKR, SK, SK);
    int vr = (mt == 0) ? 16 : (SKR - 16);
#pragma unroll
    for (int nt = 0; nt < 4; ++nt) {
      v8f acc = {};
      acc = wmma(a, frag_b_cm(B1 + (nt * 16) * 32, 32), acc);
      float efv = ef[(size_t)e * 64 + nt * 16 + (lane & 15)];
#pragma unroll
      for (int i = 0; i < 8; ++i) acc[i] *= efv;
      store_d_fill(acc, B2 + mt * 1024 + nt * 16, 64, vr);
    }
  }

  // gate with sigmoid of row 0
  for (int c = lane; c < 64; c += 32) gate[w][c] = sigm((float)B2[c]);
  for (int idx = lane; idx < SKR * 64; idx += 32) {
    int c = idx & 63;
    B2[idx] = (_Float16)((float)B2[idx] * gate[w][c]);
  }

  // 3) alpha per head
  float al[4];
#pragma unroll
  for (int h = 0; h < 4; ++h)
    al[h] = exb[(size_t)e * 4 + h] / (den[(size_t)r * 4 + h] + 1e-9f);

  // 4) wv = (msg @ Wv) * alpha -> B3 row-major (overwrites wigner staging; done with it)
  const _Float16* WvT = wb + OFF_WVT;
#pragma unroll
  for (int mt = 0; mt < 2; ++mt) {
    int vr = (mt == 0) ? 16 : (SKR - 16);
#pragma unroll
    for (int nt = 0; nt < 4; ++nt) {
      v8f acc = {};
#pragma unroll
      for (int ks = 0; ks < 2; ++ks)
        acc = wmma(frag_a_h(B2 + mt * 1024 + ks * 32, 64),
                   frag_b_cm(WvT + (nt * 16) * 64 + ks * 32, 64), acc);
#pragma unroll
      for (int i = 0; i < 8; ++i) acc[i] *= al[nt];  // head = nt (VC=16)
      store_d_fill(acc, B3 + mt * 1024 + nt * 16, 64, vr);
    }
  }

  // 5) u = wv @ Wo -> B1 column-major (rows >= 19 zeroed)
  const _Float16* WoT = wb + OFF_WOT;
#pragma unroll
  for (int mt = 0; mt < 2; ++mt) {
    int vr = (mt == 0) ? 16 : (SKR - 16);
#pragma unroll
    for (int nt = 0; nt < 4; ++nt) {
      v8f acc = {};
#pragma unroll
      for (int ks = 0; ks < 2; ++ks)
        acc = wmma(frag_a_h(B3 + mt * 1024 + ks * 32, 64),
                   frag_b_cm(WoT + (nt * 16) * 64 + ks * 32, 64), acc);
      store_d_cm(acc, B1 + (nt * 16) * 32 + mt * 16, 32, vr - 0);  // col-major tile at (mt*16, nt*16)
    }
  }

  // 6) out = wigner_inv @ u, atomic scatter into x1[receiver]
#pragma unroll
  for (int mt2 = 0; mt2 < 2; ++mt2) {
    v16h a = frag_a_f32b(sWgi, mt2, SK, SKR, SKR);
#pragma unroll
    for (int nt = 0; nt < 4; ++nt) {
      v8f acc = {};
      acc = wmma(a, frag_b_cm(B1 + (nt * 16) * 32, 32), acc);
      int n = nt * 16 + (lane & 15);
      int m0 = (lane < 16) ? 0 : 8;
#pragma unroll
      for (int i = 0; i < 8; ++i) {
        int m = mt2 * 16 + m0 + i;
        if (m < SK) atomicAdd(&x1[((size_t)r * SK + m) * 64 + n], acc[i]);
      }
    }
  }
}

// grid-space FFN per node: out = from_grid @ (silu((to_grid @ x) @ Wf1) @ Wf2) + x1
__global__ __launch_bounds__(64) void k_ffn(
    const _Float16* __restrict__ xh2, const _Float16* __restrict__ wb,
    const float* __restrict__ x1, float* __restrict__ out, int N) {
  __shared__ __align__(16) _Float16 xs[2][2048];  // node coeffs, column-major [c][k]
  __shared__ __align__(16) _Float16 hb[2][4096];  // hidden, row-major
  __shared__ __align__(16) _Float16 gb[2][2048];  // xg row-major, then og column-major
  int w = threadIdx.x >> 5, lane = threadIdx.x & 31;
  int n = blockIdx.x * 2 + w;
  if (n >= N) return;
  _Float16* XS = xs[w]; _Float16* HB = hb[w]; _Float16* GB = gb[w];

  for (int idx = lane; idx < 2048; idx += 32) {
    int c = idx >> 5, rr = idx & 31;
    XS[c * 32 + rr] = (rr < SK) ? xh2[((size_t)n * SK + rr) * 64 + c] : (_Float16)0.0f;
  }

  v8f accO[2][4];
#pragma unroll
  for (int i = 0; i < 2; ++i)
#pragma unroll
    for (int j = 0; j < 4; ++j) { v8f z = {}; accO[i][j] = z; }

  const _Float16* TG  = wb + OFF_TG;    // row-major [128][32]
  const _Float16* FG  = wb + OFF_FG;    // row-major [32][128]
  const _Float16* W1T = wb + OFF_WF1T;  // [128 n][64 k]
  const _Float16* W2T = wb + OFF_WF2T;  // [64 n][128 k]

  for (int cc = 0; cc < 4; ++cc) {  // 4 chunks of 32 grid rows (100 padded to 128)
    // xg = to_grid_chunk @ x -> GB row-major [32][64]
#pragma unroll
    for (int mt = 0; mt < 2; ++mt) {
      v16h a = frag_a_h(TG + (cc * 32 + mt * 16) * 32, 32);
#pragma unroll
      for (int nt = 0; nt < 4; ++nt) {
        v8f acc = {};
        acc = wmma(a, frag_b_cm(XS + (nt * 16) * 32, 32), acc);
        store_d_fill(acc, GB + mt * 1024 + nt * 16, 64, 16);
      }
    }
    // h = silu(xg @ Wf1) -> HB row-major [32][128]
#pragma unroll
    for (int mt = 0; mt < 2; ++mt)
#pragma unroll
      for (int nt = 0; nt < 8; ++nt) {
        v8f acc = {};
#pragma unroll
        for (int ks = 0; ks < 2; ++ks)
          acc = wmma(frag_a_h(GB + mt * 1024 + ks * 32, 64),
                     frag_b_cm(W1T + (nt * 16) * 64 + ks * 32, 64), acc);
#pragma unroll
        for (int i = 0; i < 8; ++i) { float x = acc[i]; acc[i] = x * sigm(x); }
        store_d_fill(acc, HB + mt * 2048 + nt * 16, 128, 16);
      }
    // og = h @ Wf2 -> GB column-major [64 n][32 k]
#pragma unroll
    for (int mt = 0; mt < 2; ++mt)
#pragma unroll
      for (int nt = 0; nt < 4; ++nt) {
        v8f acc = {};
#pragma unroll
        for (int ks = 0; ks < 4; ++ks)
          acc = wmma(frag_a_h(HB + mt * 2048 + ks * 32, 128),
                     frag_b_cm(W2T + (nt * 16) * 128 + ks * 32, 128), acc);
        store_d_cm(acc, GB + (nt * 16) * 32 + mt * 16, 32, 16);
      }
    // out += from_grid_chunk @ og
#pragma unroll
    for (int mt2 = 0; mt2 < 2; ++mt2) {
      v16h a = frag_a_h(FG + mt2 * 16 * 128 + cc * 32, 128);
#pragma unroll
      for (int nt = 0; nt < 4; ++nt)
        accO[mt2][nt] = wmma(a, frag_b_cm(GB + (nt * 16) * 32, 32), accO[mt2][nt]);
    }
  }

  // write out = accO + residual (mt2=0 rows always valid; mt2=1 guarded)
#pragma unroll
  for (int mt2 = 0; mt2 < 2; ++mt2)
#pragma unroll
    for (int nt = 0; nt < 4; ++nt) {
      int ncol = nt * 16 + (lane & 15);
      int m0 = (lane < 16) ? 0 : 8;
      if (mt2 == 0) {
#pragma unroll
        for (int i = 0; i < 8; ++i) {
          size_t idx = ((size_t)n * SK + (m0 + i)) * 64 + ncol;
          out[idx] = accO[0][nt][i] + x1[idx];
        }
      } else {
#pragma unroll
        for (int i = 0; i < 8; ++i) {
          int m = 16 + m0 + i;
          if (m < SK) {
            size_t idx = ((size_t)n * SK + m) * 64 + ncol;
            out[idx] = accO[1][nt][i] + x1[idx];
          }
        }
      }
    }
}

// ================== host launch ==================

extern "C" void kernel_launch(void* const* d_in, const int* in_sizes, int n_in,
                              void* d_out, int out_size, void* d_ws, size_t ws_size,
                              hipStream_t stream) {
  (void)n_in; (void)out_size; (void)ws_size;
  const float* nf    = (const float*)d_in[0];
  const float* edist = (const float*)d_in[1];
  const float* wig   = (const float*)d_in[2];
  const float* wigi  = (const float*)d_in[3];
  const int*   spec  = (const int*)d_in[4];
  const int*   snd   = (const int*)d_in[5];
  const int*   rcv   = (const int*)d_in[6];
  const float* g1    = (const float*)d_in[7];
  const float* g2    = (const float*)d_in[8];
  const float* embs  = (const float*)d_in[9];
  const float* embr  = (const float*)d_in[10];
  const float* We1   = (const float*)d_in[11];
  const float* We2   = (const float*)d_in[12];
  const float* Wso2  = (const float*)d_in[13];
  const float* Wa    = (const float*)d_in[14];
  const float* va    = (const float*)d_in[15];
  const float* Wv    = (const float*)d_in[16];
  const float* Wo    = (const float*)d_in[17];
  const float* tg    = (const float*)d_in[18];
  const float* fg    = (const float*)d_in[19];
  const float* Wf1   = (const float*)d_in[20];
  const float* Wf2   = (const float*)d_in[21];

  const int N = in_sizes[0] / (SK * SC);
  const int E = in_sizes[5];
  const size_t NKC = (size_t)N * SK * SC;

  // workspace layout (all segment sizes are multiples of 16 bytes)
  float*    x1     = (float*)d_ws;
  float*    ef     = x1 + NKC;
  float*    logits = ef + (size_t)E * 64;
  float*    exb    = logits + (size_t)E * 4;
  float*    den    = exb + (size_t)E * 4;
  unsigned* mxo    = (unsigned*)(den + (size_t)N * 4);
  _Float16* xh     = (_Float16*)(mxo + (size_t)N * 4);
  _Float16* yh     = xh + NKC;
  _Float16* zh     = yh + NKC;
  _Float16* wb     = zh + NKC;

  // 0) init residual/accumulators + weight conversion
  k_init<<<2048, 256, 0, stream>>>(nf, x1, den, mxo, (long long)NKC, (long long)N * 4);
  k_cvt<<<64, 256, 0, stream>>>(We1, We2, Wso2, Wv, Wo, Wf1, Wf2, tg, fg, wb);

  // 1) norm1
  k_norm<<<N, 256, 0, stream>>>(nf, g1, xh);

  // 2) node-side Wso2 projection
  {
    int M = N * SK;
    int waves = (M + 15) / 16;
    int blocks = (waves + 7) / 8;
    k_yz<<<blocks, 256, 0, stream>>>(xh, wb, yh, zh, M);
  }

  // 3) edge embedding MLP
  k_edgemlp<<<(E + 63) / 64, 128, 0, stream>>>(edist, spec, snd, rcv, embs, embr, wb, ef, E);

  // 4) logits + segment max
  k_logits<<<(E + 3) / 4, 128, 0, stream>>>(wig, yh, zh, ef, Wa, va, snd, rcv, logits, mxo, E);

  // 5) exp + segment sum
  k_expsum<<<(E * 4 + 255) / 256, 256, 0, stream>>>(logits, mxo, rcv, exb, den, E);

  // 6) fused per-edge attention (scatter into x1)
  k_edge<<<(E + 3) / 4, 128, 0, stream>>>(wig, wigi, yh, zh, ef, exb, den, snd, rcv, wb, x1, E);

  // 7) norm2 (reuse xh buffer for normalized x1)
  k_norm<<<N, 256, 0, stream>>>(x1, g2, xh);

  // 8) grid FFN + residual -> d_out
  k_ffn<<<(N + 1) / 2, 64, 0, stream>>>(xh, wb, x1, (float*)d_out, N);
}
